// InfoNCE_24464133718914
// MI455X (gfx1250) — matrix operand order repaired and checked by reading, not compiled, not run on hardware
//
#include <hip/hip_runtime.h>
#include <hip/hip_bf16.h>

typedef __attribute__((ext_vector_type(16))) _Float16 v16h;
typedef __attribute__((ext_vector_type(8)))  _Float16 v8h;
typedef __attribute__((ext_vector_type(8)))  float    v8f;
typedef unsigned int u32x4 __attribute__((ext_vector_type(4)));
typedef int          i32x4 __attribute__((ext_vector_type(4)));
typedef int          i32x8 __attribute__((ext_vector_type(8)));

#define NROWS 4096   // 2*BATCH
#define BATCH 2048
#define DIM   256
#define INV_T 2.0f   // 1/TEMP, TEMP = 0.5
#define LDS_ROW 264  // 256 f16 payload + 8 f16 pad -> 528B row stride, 16B aligned

// ---------------------------------------------------------------------------
// Kernel 1: L2-normalize rows of [emb_i; emb_j] -> f16 z matrix + inv norms
// ---------------------------------------------------------------------------
__global__ __launch_bounds__(256) void nce_normalize(
    const float* __restrict__ emb_i, const float* __restrict__ emb_j,
    _Float16* __restrict__ zh, float* __restrict__ invn) {
  const int row = blockIdx.x;            // 0..4095
  const int t   = threadIdx.x;           // 0..255
  const float* src = (row < BATCH) ? (emb_i + row * DIM)
                                   : (emb_j + (row - BATCH) * DIM);
  const float x = src[t];
  float s = x * x;
  #pragma unroll
  for (int m = 1; m < 32; m <<= 1) s += __shfl_xor(s, m, 32);
  __shared__ float wsum[8];
  const int lane = t & 31, wv = t >> 5;
  if (lane == 0) wsum[wv] = s;
  __syncthreads();
  float tot = wsum[0];
  #pragma unroll
  for (int i = 1; i < 8; ++i) tot += wsum[i];
  const float inv = 1.0f / fmaxf(sqrtf(tot), 1e-12f);
  zh[row * DIM + t] = (_Float16)(x * inv);
  if (t == 0) invn[row] = inv;
}

// ---------------------------------------------------------------------------
// Kernel 2: exact-f32 positives: pos[i] = <emb_i[i], emb_j[i]> * invn_i*invn_j
// ---------------------------------------------------------------------------
__global__ __launch_bounds__(256) void nce_positives(
    const float* __restrict__ emb_i, const float* __restrict__ emb_j,
    const float* __restrict__ invn, float* __restrict__ pos) {
  const int i = blockIdx.x;              // 0..2047
  const int t = threadIdx.x;
  float s = emb_i[i * DIM + t] * emb_j[i * DIM + t];
  #pragma unroll
  for (int m = 1; m < 32; m <<= 1) s += __shfl_xor(s, m, 32);
  __shared__ float wsum[8];
  const int lane = t & 31, wv = t >> 5;
  if (lane == 0) wsum[wv] = s;
  __syncthreads();
  float tot = wsum[0];
  #pragma unroll
  for (int i2 = 1; i2 < 8; ++i2) tot += wsum[i2];
  if (t == 0) pos[i] = tot * invn[i] * invn[BATCH + i];
}

// ---------------------------------------------------------------------------
// Kernel 3: fused WMMA Gram-matrix + masked exp row-sum, TDM-staged B tiles.
// grid = (32 row-blocks, 8 col-slices), block = 256 threads = 8 waves.
// All 8 waves share each 16-col x 256-K B tile: wave 0 DMAs it into LDS with
// TENSOR_LOAD_TO_LDS (double-buffered, TENSORcnt-tracked), everyone consumes
// via ds_load_b128 + v_wmma_f32_16x16x32_f16, then exp-accumulates row sums.
// ---------------------------------------------------------------------------
__global__ __launch_bounds__(256) void nce_gemm_rowsum(
    const _Float16* __restrict__ zh, float* __restrict__ rowsum_part) {
  __shared__ __align__(16) _Float16 lds_tile[2][16 * LDS_ROW];

  const int lane  = threadIdx.x & 31;
  const int wave  = threadIdx.x >> 5;
  const int h     = lane >> 4;           // half-wave select
  const int l16   = lane & 15;
  const int mBase = (blockIdx.x * 8 + wave) * 16;
  const int m     = mBase + l16;
  const int colBase = blockIdx.y * 512;

  // TDM issue: DMA 16 rows x 512B of zh (rows nBase..nBase+15) into LDS buffer
  // b, padding each 128-DWORD row with 4 DWORDs (-> 528B LDS row stride).
  auto tdm_issue = [&](int ct, int b) {
    const unsigned long long ga =
        (unsigned long long)(uintptr_t)zh +
        (unsigned long long)(colBase + ct * 16) * (DIM * 2);
    const unsigned la = (unsigned)(uintptr_t)(&lds_tile[b][0]); // flat->LDS low bits
    u32x4 g0;
    g0[0] = 1u;                                   // count=1, load, no gather
    g0[1] = la;                                   // lds_addr [63:32]
    g0[2] = (unsigned)ga;                         // global_addr [95:64]
    g0[3] = (unsigned)((ga >> 32) & 0x1FFFFFFu)   // global_addr [120:96]
          | 0x80000000u;                          // type=2 ("image") [127:126]
    i32x8 g1;
    g1[0] = 0x07930000;   // data_size=8B | pad_enable | pad_interval=128DW | pad_amount=4DW
    g1[1] = 64 << 16;     // tensor_dim0 = 64 (x8B = 512B row)
    g1[2] = 16 << 16;     // tensor_dim1 = 16 rows
    g1[3] = 64 << 16;     // tile_dim0 = 64
    g1[4] = 16;           // tile_dim1 = 16, tile_dim2 = 0
    g1[5] = 64;           // tensor_dim0_stride = 64 (x8B = 512B)
    g1[6] = 0;
    g1[7] = 0;
    i32x4 gz4 = {0, 0, 0, 0};
    i32x8 gz8 = {0, 0, 0, 0, 0, 0, 0, 0};
    __builtin_amdgcn_tensor_load_to_lds(g0, g1, gz4, gz4, gz8, 0);
  };

  // A fragments (held in regs for whole kernel): 16-bit A 16x32 layout.
  v16h afrag[8];
  const _Float16* arow = zh + m * DIM;
  #pragma unroll
  for (int kt = 0; kt < 8; ++kt) {
    const _Float16* p = arow + kt * 32 + h * 8;
    v8h lo = *(const v8h*)(p);           // K = kt*32 + h*8 .. +7
    v8h hi = *(const v8h*)(p + 16);      // K = kt*32 + 16 + h*8 .. +7
    afrag[kt] = __builtin_shufflevector(lo, hi,
        0, 1, 2, 3, 4, 5, 6, 7, 8, 9, 10, 11, 12, 13, 14, 15);
  }

  if (wave == 0) tdm_issue(0, 0);        // prologue: tile 0 -> buffer 0

  float partial[8] = {0.f, 0.f, 0.f, 0.f, 0.f, 0.f, 0.f, 0.f};

  for (int ct = 0; ct < 32; ++ct) {
    const int cur = ct & 1;
    if (wave == 0) {
      if (ct + 1 < 32) {
        tdm_issue(ct + 1, cur ^ 1);                // prefetch next tile
        __builtin_amdgcn_s_wait_tensorcnt(1);      // in-order: tile ct done
      } else {
        __builtin_amdgcn_s_wait_tensorcnt(0);
      }
    }
    __syncthreads();                               // buf[cur] valid for all waves

    const int nBase = colBase + ct * 16;
    // B fragment from LDS: lane = column n (row l16 of tile); per lane K is
    // K = kt*32 + h*8 .. +7, then kt*32 + 16 + h*8 .. +7.
    v8f acc = {};
    #pragma unroll
    for (int kt = 0; kt < 8; ++kt) {
      const int e = l16 * LDS_ROW + kt * 32 + h * 8;
      v8h lo = *(const v8h*)(&lds_tile[cur][e]);
      v8h hi = *(const v8h*)(&lds_tile[cur][e + 16]);
      v16h bfrag = __builtin_shufflevector(lo, hi,
          0, 1, 2, 3, 4, 5, 6, 7, 8, 9, 10, 11, 12, 13, 14, 15);
      acc = __builtin_amdgcn_wmma_f32_16x16x32_f16(
          false, afrag[kt], false, bfrag, (short)0, acc, false, false);
    }
    // Epilogue: exp(sim / T) with diagonal (j == i) masked.
    #pragma unroll
    for (int r = 0; r < 8; ++r) {
      const int mr = mBase + h * 8 + r;
      const int n  = nBase + l16;
      const float e = __expf(acc[r] * INV_T);
      partial[r] += (mr == n) ? 0.0f : e;
    }
    __syncthreads();                               // done with buf[cur] before overwrite
  }

  // Reduce over the 16 lanes sharing the same row (within each half-wave).
  #pragma unroll
  for (int r = 0; r < 8; ++r) {
    float v = partial[r];
    v += __shfl_xor(v, 1, 16);
    v += __shfl_xor(v, 2, 16);
    v += __shfl_xor(v, 4, 16);
    v += __shfl_xor(v, 8, 16);
    if (l16 == 0)
      rowsum_part[(mBase + h * 8 + r) * 8 + blockIdx.y] = v;
  }
}

// ---------------------------------------------------------------------------
// Kernel 4: loss = mean_i( log(denom_i) - pos_i / T ), single block.
// ---------------------------------------------------------------------------
__global__ __launch_bounds__(256) void nce_loss(
    const float* __restrict__ rowsum_part, const float* __restrict__ pos,
    float* __restrict__ out) {
  const int t = threadIdx.x;
  float acc = 0.0f;
  for (int k = 0; k < 16; ++k) {
    const int row = t * 16 + k;          // 0..4095
    float denom = 0.0f;
    #pragma unroll
    for (int s = 0; s < 8; ++s) denom += rowsum_part[row * 8 + s];
    const int pi = (row < BATCH) ? row : row - BATCH;
    acc += logf(denom) - INV_T * pos[pi];
  }
  __shared__ float red[256];
  red[t] = acc;
  __syncthreads();
  for (int s = 128; s > 0; s >>= 1) {
    if (t < s) red[t] += red[t + s];
    __syncthreads();
  }
  if (t == 0) out[0] = red[0] / (float)NROWS;
}

// ---------------------------------------------------------------------------
extern "C" void kernel_launch(void* const* d_in, const int* in_sizes, int n_in,
                              void* d_out, int out_size, void* d_ws, size_t ws_size,
                              hipStream_t stream) {
  const float* emb_i = (const float*)d_in[0];
  const float* emb_j = (const float*)d_in[1];
  float* out = (float*)d_out;

  // Workspace layout:
  //   [0, 2MB)            : f16 z matrix, 4096 x 256
  //   then floats         : invn[4096], pos[2048], rowsum_part[4096*8]
  _Float16* zh = (_Float16*)d_ws;
  float* fws = (float*)((char*)d_ws + (size_t)NROWS * DIM * sizeof(_Float16));
  float* invn        = fws;
  float* pos         = fws + NROWS;
  float* rowsum_part = fws + NROWS + BATCH;

  nce_normalize<<<NROWS, 256, 0, stream>>>(emb_i, emb_j, zh, invn);
  nce_positives<<<BATCH, 256, 0, stream>>>(emb_i, emb_j, invn, pos);
  nce_gemm_rowsum<<<dim3(32, 8), 256, 0, stream>>>(zh, rowsum_part);
  nce_loss<<<1, 256, 0, stream>>>(rowsum_part, pos, out);
}
